// VertexGNN_86990267613952
// MI455X (gfx1250) — matrix operand order repaired and checked by reading, not compiled, not run on hardware
//
#include <hip/hip_runtime.h>
#include <hip/hip_bf16.h>
#include <math.h>

typedef __attribute__((ext_vector_type(16))) _Float16 v16h;
typedef __attribute__((ext_vector_type(8)))  float    v8f;

#define LNEPS 1e-5f

__device__ __forceinline__ float gelu_exact(float x) {
    return 0.5f * x * (1.0f + erff(x * 0.70710678118654752f));
}

__device__ __forceinline__ void pack8(v16h& a, int base, float4 p, float4 q) {
    a[base + 0] = (_Float16)p.x; a[base + 1] = (_Float16)p.y;
    a[base + 2] = (_Float16)p.z; a[base + 3] = (_Float16)p.w;
    a[base + 4] = (_Float16)q.x; a[base + 5] = (_Float16)q.y;
    a[base + 6] = (_Float16)q.z; a[base + 7] = (_Float16)q.w;
}

// ---------------------------------------------------------------- zero
__global__ void k_zero(float* p, size_t n) {
    size_t i = (size_t)blockIdx.x * blockDim.x + threadIdx.x;
    size_t stride = (size_t)gridDim.x * blockDim.x;
    for (; i < n; i += stride) p[i] = 0.0f;
}

// ---------------------------------------------------------------- degree
__global__ void k_degree(const int* dst, float* deg, int E) {
    int i = blockIdx.x * blockDim.x + threadIdx.x;
    int stride = gridDim.x * blockDim.x;
    for (; i < E; i += stride) atomicAdd(&deg[dst[i]], 1.0f);
}

// ---------------------------------------------------------------- node embed
__global__ void k_node_embed(const float* __restrict__ x,
                             const float* __restrict__ w1, const float* __restrict__ b1,
                             const float* __restrict__ g1, const float* __restrict__ be1,
                             const float* __restrict__ w2, const float* __restrict__ b2,
                             const float* __restrict__ g2, const float* __restrict__ be2,
                             float* __restrict__ h, int N) {
    int gwid  = (blockIdx.x * blockDim.x + threadIdx.x) >> 5;
    int lane  = threadIdx.x & 31;
    int nwave = (gridDim.x * blockDim.x) >> 5;
    for (int n = gwid; n < N; n += nwave) {
        float x0 = x[(size_t)n * 2 + 0];
        float x1 = x[(size_t)n * 2 + 1];
        int f0 = lane, f1 = lane + 32;
        float t0 = x0 * w1[f0] + x1 * w1[64 + f0] + b1[f0];
        float t1 = x0 * w1[f1] + x1 * w1[64 + f1] + b1[f1];
        float s = t0 + t1, s2 = t0 * t0 + t1 * t1;
        #pragma unroll
        for (int off = 1; off < 32; off <<= 1) { s += __shfl_xor(s, off, 32); s2 += __shfl_xor(s2, off, 32); }
        float mu = s * (1.0f / 64.0f);
        float rstd = rsqrtf(s2 * (1.0f / 64.0f) - mu * mu + LNEPS);
        float u0 = gelu_exact((t0 - mu) * rstd * g1[f0] + be1[f0]);
        float u1 = gelu_exact((t1 - mu) * rstd * g1[f1] + be1[f1]);
        float a0 = b2[f0], a1 = b2[f1];
        #pragma unroll
        for (int k = 0; k < 64; ++k) {
            float uk = __shfl((k < 32) ? u0 : u1, k & 31, 32);
            a0 += uk * w2[k * 64 + f0];
            a1 += uk * w2[k * 64 + f1];
        }
        s = a0 + a1; s2 = a0 * a0 + a1 * a1;
        #pragma unroll
        for (int off = 1; off < 32; off <<= 1) { s += __shfl_xor(s, off, 32); s2 += __shfl_xor(s2, off, 32); }
        mu = s * (1.0f / 64.0f);
        rstd = rsqrtf(s2 * (1.0f / 64.0f) - mu * mu + LNEPS);
        h[(size_t)n * 64 + f0] = gelu_exact((a0 - mu) * rstd * g2[f0] + be2[f0]);
        h[(size_t)n * 64 + f1] = gelu_exact((a1 - mu) * rstd * g2[f1] + be2[f1]);
    }
}

// ---------------------------------------------------------------- edge message (WMMA)
// Wave owns a 16-edge tile. K=0..63 (the h[src] features) runs on the WMMA pipe
// with register-resident B fragments; the K=64,65 edge_attr columns are a rank-2
// VALU update co-executing with the WMMAs. No LDS, no barriers, no spills.
__global__ void __launch_bounds__(256, 1)
k_edge_msg(const float* __restrict__ h,
           const float* __restrict__ edge_attr,
           const int* __restrict__ src, const int* __restrict__ dst,
           const float* __restrict__ conv_w, const float* __restrict__ conv_b,
           const float* __restrict__ conv_mg, const float* __restrict__ conv_mb,
           float* __restrict__ agg, int E, int layer) {
    int tid = threadIdx.x, lane = tid & 31;
    int nl = lane & 15;
    const float* W = conv_w + (size_t)layer * 66 * 64;

    // B fragments (K 0..63), resident in VGPRs for the whole kernel.
    // Fragment (ks,nt): lane holds column n = nt*16+nl, K = ks*32 + kb + j.
    int kb = (lane & 16) ? 16 : 0;
    v16h B0[4], B1[4];
    float w64_r[4], w65_r[4], bias_r[4], mg_r[4], mb_r[4];
    #pragma unroll
    for (int nt = 0; nt < 4; ++nt) {
        int n = nt * 16 + nl;
        #pragma unroll
        for (int j = 0; j < 16; ++j) {
            B0[nt][j] = (_Float16)W[(size_t)(kb + j) * 64 + n];
            B1[nt][j] = (_Float16)W[(size_t)(32 + kb + j) * 64 + n];
        }
        w64_r[nt]  = W[(size_t)64 * 64 + n];
        w65_r[nt]  = W[(size_t)65 * 64 + n];
        bias_r[nt] = conv_b[layer * 64 + n];
        mg_r[nt]   = conv_mg[layer * 64 + n];
        mb_r[nt]   = conv_mb[layer * 64 + n];
    }

    int gw = (blockIdx.x * blockDim.x + tid) >> 5;
    int nw = (gridDim.x * blockDim.x) >> 5;
    int T  = (E + 15) >> 4;
    int koff = (lane & 16) ? 16 : 0;  // A K-offset for this lane group
    int mofs = (lane & 16) ? 8 : 0;   // C/D row offset for this lane group

    for (int tile = gw; tile < T; tile += nw) {
        int e0 = tile << 4;
        int em = e0 + nl;                       // edge whose A-row this lane holds
        int ec = (em < E) ? em : (E - 1);       // clamped: loads always valid
        int s  = src[ec];
        const float* hp = h + (size_t)s * 64;

        // A fragments ks=0 (K 0..31), ks=1 (K 32..63) straight from global h
        v16h a0, a1;
        {
            float4 r0 = *(const float4*)(hp + koff);
            float4 r1 = *(const float4*)(hp + koff + 4);
            float4 r2 = *(const float4*)(hp + koff + 8);
            float4 r3 = *(const float4*)(hp + koff + 12);
            pack8(a0, 0, r0, r1); pack8(a0, 8, r2, r3);
            r0 = *(const float4*)(hp + 32 + koff);
            r1 = *(const float4*)(hp + 32 + koff + 4);
            r2 = *(const float4*)(hp + 32 + koff + 8);
            r3 = *(const float4*)(hp + 32 + koff + 12);
            pack8(a1, 0, r0, r1); pack8(a1, 8, r2, r3);
        }
        float ea0 = edge_attr[(size_t)ec * 2 + 0];   // lanes 0..15 hold rows 0..15
        float ea1 = edge_attr[(size_t)ec * 2 + 1];

        v8f c[4];
        #pragma unroll
        for (int nt = 0; nt < 4; ++nt) c[nt] = (v8f){};

        #pragma unroll
        for (int nt = 0; nt < 4; ++nt)
            c[nt] = __builtin_amdgcn_wmma_f32_16x16x32_f16(false, a0, false, B0[nt], (short)0, c[nt], false, false);
        #pragma unroll
        for (int nt = 0; nt < 4; ++nt)
            c[nt] = __builtin_amdgcn_wmma_f32_16x16x32_f16(false, a1, false, B1[nt], (short)0, c[nt], false, false);

        // rank-2 edge_attr update + bias
        #pragma unroll
        for (int v = 0; v < 8; ++v) {
            int m = mofs + v;
            float e0v = __shfl(ea0, m, 32);
            float e1v = __shfl(ea1, m, 32);
            #pragma unroll
            for (int nt = 0; nt < 4; ++nt)
                c[nt][v] += bias_r[nt] + e0v * w64_r[nt] + e1v * w65_r[nt];
        }

        // per-row LN + exact GELU + atomic scatter into agg[dst]
        #pragma unroll
        for (int v = 0; v < 8; ++v) {
            int e = e0 + mofs + v;
            float s1 = c[0][v] + c[1][v] + c[2][v] + c[3][v];
            float s2 = c[0][v]*c[0][v] + c[1][v]*c[1][v] + c[2][v]*c[2][v] + c[3][v]*c[3][v];
            #pragma unroll
            for (int off = 1; off < 16; off <<= 1) {   // reduce within 16-lane half-group
                s1 += __shfl_xor(s1, off, 32);
                s2 += __shfl_xor(s2, off, 32);
            }
            float mu   = s1 * (1.0f / 64.0f);
            float rstd = rsqrtf(s2 * (1.0f / 64.0f) - mu * mu + LNEPS);
            if (e < E) {
                float* ar = agg + (size_t)dst[e] * 64;
                #pragma unroll
                for (int nt = 0; nt < 4; ++nt) {
                    int n = nt * 16 + nl;
                    float xx = (c[nt][v] - mu) * rstd * mg_r[nt] + mb_r[nt];
                    atomicAdd(ar + n, gelu_exact(xx));
                }
            }
        }
    }
}

// ---------------------------------------------------------------- node update
__global__ void k_node_update(float* __restrict__ h, const float* __restrict__ agg,
                              const float* __restrict__ deg,
                              const float* __restrict__ ng, const float* __restrict__ nb,
                              const float* __restrict__ og, const float* __restrict__ ob, int N) {
    int gwid  = (blockIdx.x * blockDim.x + threadIdx.x) >> 5;
    int lane  = threadIdx.x & 31;
    int nwave = (gridDim.x * blockDim.x) >> 5;
    for (int n = gwid; n < N; n += nwave) {
        float r = 1.0f / fmaxf(deg[n], 1.0f);
        int f0 = lane, f1 = lane + 32;
        size_t base = (size_t)n * 64;
        float h0 = h[base + f0], h1 = h[base + f1];
        float a0 = agg[base + f0] * r + h0;
        float a1 = agg[base + f1] * r + h1;
        float s = a0 + a1, s2 = a0 * a0 + a1 * a1;
        #pragma unroll
        for (int off = 1; off < 32; off <<= 1) { s += __shfl_xor(s, off, 32); s2 += __shfl_xor(s2, off, 32); }
        float mu = s * (1.0f / 64.0f);
        float rstd = rsqrtf(s2 * (1.0f / 64.0f) - mu * mu + LNEPS);
        float c0 = (a0 - mu) * rstd * ng[f0] + nb[f0];
        float c1 = (a1 - mu) * rstd * ng[f1] + nb[f1];
        s = c0 + c1; s2 = c0 * c0 + c1 * c1;
        #pragma unroll
        for (int off = 1; off < 32; off <<= 1) { s += __shfl_xor(s, off, 32); s2 += __shfl_xor(s2, off, 32); }
        mu = s * (1.0f / 64.0f);
        rstd = rsqrtf(s2 * (1.0f / 64.0f) - mu * mu + LNEPS);
        h[base + f0] = h0 + (c0 - mu) * rstd * og[f0] + ob[f0];
        h[base + f1] = h1 + (c1 - mu) * rstd * og[f1] + ob[f1];
    }
}

// ---------------------------------------------------------------- global mean pool
__global__ void k_pool(const float* __restrict__ h, const int* __restrict__ batch,
                       float* __restrict__ gsum, float* __restrict__ cnt, int N) {
    int gwid  = (blockIdx.x * blockDim.x + threadIdx.x) >> 5;
    int lane  = threadIdx.x & 31;
    int nwave = (gridDim.x * blockDim.x) >> 5;
    for (int n = gwid; n < N; n += nwave) {
        int b = batch[n];
        size_t base = (size_t)n * 64;
        atomicAdd(&gsum[(size_t)b * 64 + lane],      h[base + lane]);
        atomicAdd(&gsum[(size_t)b * 64 + lane + 32], h[base + lane + 32]);
        if (lane == 0) atomicAdd(&cnt[b], 1.0f);
    }
}

// ---------------------------------------------------------------- head MLP (wave per graph)
__global__ void k_head(const float* __restrict__ gsum, const float* __restrict__ cnt,
                       const float* __restrict__ w1, const float* __restrict__ b1,
                       const float* __restrict__ g1, const float* __restrict__ be1,
                       const float* __restrict__ w2, const float* __restrict__ b2,
                       const float* __restrict__ g2, const float* __restrict__ be2,
                       const float* __restrict__ w3, const float* __restrict__ b3,
                       float* __restrict__ out, int B) {
    int gwid  = (blockIdx.x * blockDim.x + threadIdx.x) >> 5;
    int lane  = threadIdx.x & 31;
    int nwave = (gridDim.x * blockDim.x) >> 5;
    for (int b = gwid; b < B; b += nwave) {
        float r  = 1.0f / fmaxf(cnt[b], 1.0f);
        int f0 = lane, f1 = lane + 32;
        float x0 = gsum[(size_t)b * 64 + f0] * r;
        float x1 = gsum[(size_t)b * 64 + f1] * r;
        float a0 = b1[f0], a1 = b1[f1];
        #pragma unroll
        for (int k = 0; k < 64; ++k) {
            float xk = __shfl((k < 32) ? x0 : x1, k & 31, 32);
            a0 += xk * w1[k * 64 + f0];
            a1 += xk * w1[k * 64 + f1];
        }
        float s = a0 + a1, s2 = a0 * a0 + a1 * a1;
        #pragma unroll
        for (int off = 1; off < 32; off <<= 1) { s += __shfl_xor(s, off, 32); s2 += __shfl_xor(s2, off, 32); }
        float mu = s * (1.0f / 64.0f);
        float rstd = rsqrtf(s2 * (1.0f / 64.0f) - mu * mu + LNEPS);
        float u0 = fmaxf((a0 - mu) * rstd * g1[f0] + be1[f0], 0.0f);
        float u1 = fmaxf((a1 - mu) * rstd * g1[f1] + be1[f1], 0.0f);
        float a = b2[lane];
        #pragma unroll
        for (int k = 0; k < 64; ++k) {
            float uk = __shfl((k < 32) ? u0 : u1, k & 31, 32);
            a += uk * w2[k * 32 + lane];
        }
        s = a; s2 = a * a;
        #pragma unroll
        for (int off = 1; off < 32; off <<= 1) { s += __shfl_xor(s, off, 32); s2 += __shfl_xor(s2, off, 32); }
        mu = s * (1.0f / 32.0f);
        rstd = rsqrtf(s2 * (1.0f / 32.0f) - mu * mu + LNEPS);
        float v = fmaxf((a - mu) * rstd * g2[lane] + be2[lane], 0.0f);
        float p = v * w3[lane];
        #pragma unroll
        for (int off = 1; off < 32; off <<= 1) p += __shfl_xor(p, off, 32);
        if (lane == 0) out[b] = p + b3[0];
    }
}

// ---------------------------------------------------------------- launcher
extern "C" void kernel_launch(void* const* d_in, const int* in_sizes, int n_in,
                              void* d_out, int out_size, void* d_ws, size_t ws_size,
                              hipStream_t stream) {
    const float* x         = (const float*)d_in[0];
    const float* edge_attr = (const float*)d_in[1];
    const float* ne_w1 = (const float*)d_in[2];
    const float* ne_b1 = (const float*)d_in[3];
    const float* ne_g1 = (const float*)d_in[4];
    const float* ne_be1= (const float*)d_in[5];
    const float* ne_w2 = (const float*)d_in[6];
    const float* ne_b2 = (const float*)d_in[7];
    const float* ne_g2 = (const float*)d_in[8];
    const float* ne_be2= (const float*)d_in[9];
    const float* conv_w  = (const float*)d_in[10];
    const float* conv_b  = (const float*)d_in[11];
    const float* conv_mg = (const float*)d_in[12];
    const float* conv_mb = (const float*)d_in[13];
    const float* conv_ng = (const float*)d_in[14];
    const float* conv_nb = (const float*)d_in[15];
    const float* out_ng  = (const float*)d_in[16];
    const float* out_nb  = (const float*)d_in[17];
    const float* m_w1 = (const float*)d_in[18];
    const float* m_b1 = (const float*)d_in[19];
    const float* m_g1 = (const float*)d_in[20];
    const float* m_be1= (const float*)d_in[21];
    const float* m_w2 = (const float*)d_in[22];
    const float* m_b2 = (const float*)d_in[23];
    const float* m_g2 = (const float*)d_in[24];
    const float* m_be2= (const float*)d_in[25];
    const float* m_w3 = (const float*)d_in[26];
    const float* m_b3 = (const float*)d_in[27];
    const int*   edge_index = (const int*)d_in[28];
    const int*   batch      = (const int*)d_in[29];

    const int N = in_sizes[0] / 2;
    const int E = in_sizes[1] / 2;
    const int B = out_size;
    const int* src = edge_index;
    const int* dst = edge_index + E;

    float* ws   = (float*)d_ws;
    float* h    = ws;                              // N*64
    float* agg  = h   + (size_t)N * 64;            // N*64
    float* deg  = agg + (size_t)N * 64;            // N
    float* gsum = deg + (size_t)N;                 // B*64
    float* cnt  = gsum + (size_t)B * 64;           // B

    size_t zn = (size_t)N + (size_t)B * 64 + (size_t)B;
    k_zero<<<1024, 256, 0, stream>>>(deg, zn);
    k_degree<<<2048, 256, 0, stream>>>(dst, deg, E);

    int node_blocks = (N + 7) / 8;
    k_node_embed<<<node_blocks, 256, 0, stream>>>(x, ne_w1, ne_b1, ne_g1, ne_be1,
                                                  ne_w2, ne_b2, ne_g2, ne_be2, h, N);

    for (int l = 0; l < 3; ++l) {
        k_zero<<<2048, 256, 0, stream>>>(agg, (size_t)N * 64);
        k_edge_msg<<<2048, 256, 0, stream>>>(h, edge_attr, src, dst,
                                             conv_w, conv_b, conv_mg, conv_mb,
                                             agg, E, l);
        k_node_update<<<node_blocks, 256, 0, stream>>>(h, agg, deg,
                                                       conv_ng + l * 64, conv_nb + l * 64,
                                                       out_ng + l * 64, out_nb + l * 64, N);
    }

    k_pool<<<node_blocks, 256, 0, stream>>>(h, batch, gsum, cnt, N);
    k_head<<<(B + 7) / 8, 256, 0, stream>>>(gsum, cnt,
                                            m_w1, m_b1, m_g1, m_be1,
                                            m_w2, m_b2, m_g2, m_be2,
                                            m_w3, m_b3, (float*)d_out, B);
}
